// DynamicEarlyExitStudent_75857712382062
// MI455X (gfx1250) — matrix operand order repaired and checked by reading, not compiled, not run on hardware
//
#include <hip/hip_runtime.h>
#include <hip/hip_bf16.h>
#include <math.h>

#define Bb 16
#define Ll 512
#define Dd 256
#define DIi 512
#define Nn 16
#define Rr 16
#define NT (Bb * Ll)   // 8192 tokens

typedef __attribute__((ext_vector_type(16))) _Float16 v16h;
typedef __attribute__((ext_vector_type(8)))  float    v8f;

__device__ __forceinline__ float sigmoidf_(float x) { return 1.f / (1.f + expf(-x)); }

__device__ __forceinline__ float block_reduce_256(float* red, int tid, float v) {
    red[tid] = v;
    __syncthreads();
    #pragma unroll
    for (int s = 128; s > 0; s >>= 1) {
        if (tid < s) red[tid] += red[tid + s];
        __syncthreads();
    }
    float r = red[0];
    __syncthreads();
    return r;
}

// ---------------- Tokenizer: embeddings + fusion matvec + LayerNorm ----------------
__global__ void tokenize_kernel(const float* __restrict__ x,
                                const float* __restrict__ emb_proto,
                                const float* __restrict__ emb_flags,
                                const float* __restrict__ emb_dir,
                                const float* __restrict__ len_w, const float* __restrict__ len_b,
                                const float* __restrict__ iat_w, const float* __restrict__ iat_b,
                                const float* __restrict__ fus_w, const float* __restrict__ fus_b,
                                const float* __restrict__ tok_g, const float* __restrict__ tok_b,
                                float* __restrict__ feats, _Float16* __restrict__ feats16) {
    __shared__ float cat[136];
    __shared__ float red[256];
    int t = blockIdx.x;          // token
    int tid = threadIdx.x;
    const float* xr = x + (size_t)t * 5;
    if (tid < 136) {
        float v;
        if (tid < 32) {
            int p = (int)xr[0]; p = p < 0 ? 0 : (p > 255 ? 255 : p);
            v = emb_proto[p * 32 + tid];
        } else if (tid < 64) {
            int f = (int)xr[2]; f = f < 0 ? 0 : (f > 63 ? 63 : f);
            v = emb_flags[f * 32 + (tid - 32)];
        } else if (tid < 72) {
            int dd = (int)xr[4]; dd = dd < 0 ? 0 : (dd > 1 ? 1 : dd);
            v = emb_dir[dd * 8 + (tid - 64)];
        } else if (tid < 104) {
            int j = tid - 72;
            v = xr[1] * len_w[j] + len_b[j];
        } else {
            int j = tid - 104;
            v = xr[3] * iat_w[j] + iat_b[j];
        }
        cat[tid] = v;
    }
    __syncthreads();
    float acc = fus_b[tid];
    const float* wr = fus_w + (size_t)tid * 136;
    #pragma unroll 8
    for (int j = 0; j < 136; ++j) acc += cat[j] * wr[j];
    float mu  = block_reduce_256(red, tid, acc) * (1.f / 256.f);
    float ex2 = block_reduce_256(red, tid, acc * acc) * (1.f / 256.f);
    float var = ex2 - mu * mu;
    float xn = (acc - mu) * rsqrtf(var + 1e-5f) * tok_g[tid] + tok_b[tid];
    feats[(size_t)t * Dd + tid] = xn;
    feats16[(size_t)t * Dd + tid] = (_Float16)xn;
}

// ---------------- WMMA GEMM: out[M,N] = A[M,K](f16) * W[N,K]^T(f16), f32 accum ----------------
// One wave computes a 16x64 strip: 1 A fragment reused across 4 N-tiles (4 independent
// WMMA accumulator chains -> hides XDL latency, cuts A-stream L2 traffic 4x).
// Fragment layouts used here (contiguous k order, not the ISA table order; 16-bit, wave32):
//   A lane(m,h): frag[0..15] = K 16h..16h+15
//   B lane(n,h): frag[0..15] = K 16h..16h+15 (contiguous along W's K-major rows)
//   D lane(n,h): VGPR r -> row r+8h, col n
__global__ void gemm_wmma_f16(const _Float16* __restrict__ A,
                              const _Float16* __restrict__ W,
                              float* __restrict__ out,
                              int M, int N, int K) {
    int lane = threadIdx.x & 31;
    int wave = blockIdx.x * (blockDim.x >> 5) + (threadIdx.x >> 5);
    int groupsN = N >> 6;            // 4 tiles (64 cols) per wave
    int tilesM = M >> 4;
    if (wave >= tilesM * groupsN) return;
    int tm = wave / groupsN;
    int tg = wave - tm * groupsN;
    int hl = lane >> 4;              // half-wave select
    int ml = lane & 15;

    const _Float16* arow = A + (size_t)(tm * 16 + ml) * K;
    const _Float16* brow = W + (size_t)(tg * 64 + ml) * K;
    v8f c0 = {}, c1 = {}, c2 = {}, c3 = {};
    for (int k0 = 0; k0 < K; k0 += 32) {
        // prefetch next K chunk of the streams (global_prefetch_b8; OOB is dropped)
        __builtin_prefetch((const void*)(arow + k0 + 128), 0, 1);
        __builtin_prefetch((const void*)(brow + k0 + 128), 0, 1);
        v16h a;
        const _Float16* ap = arow + k0 + 16 * hl;
        #pragma unroll
        for (int j = 0; j < 16; ++j) a[j] = ap[j];
        v16h b0, b1, b2, b3;
        const _Float16* bp = brow + k0 + 16 * hl;
        #pragma unroll
        for (int j = 0; j < 16; ++j) b0[j] = bp[j];
        bp += (size_t)16 * K;
        #pragma unroll
        for (int j = 0; j < 16; ++j) b1[j] = bp[j];
        bp += (size_t)16 * K;
        #pragma unroll
        for (int j = 0; j < 16; ++j) b2[j] = bp[j];
        bp += (size_t)16 * K;
        #pragma unroll
        for (int j = 0; j < 16; ++j) b3[j] = bp[j];
        c0 = __builtin_amdgcn_wmma_f32_16x16x32_f16(false, a, false, b0, (short)0, c0, false, false);
        c1 = __builtin_amdgcn_wmma_f32_16x16x32_f16(false, a, false, b1, (short)0, c1, false, false);
        c2 = __builtin_amdgcn_wmma_f32_16x16x32_f16(false, a, false, b2, (short)0, c2, false, false);
        c3 = __builtin_amdgcn_wmma_f32_16x16x32_f16(false, a, false, b3, (short)0, c3, false, false);
    }
    float* obase = out + (size_t)(tm * 16 + 8 * hl) * N + tg * 64 + ml;
    #pragma unroll
    for (int r = 0; r < 8; ++r) {
        float* orow = obase + (size_t)r * N;
        orow[0]  = c0[r];
        orow[16] = c1[r];
        orow[32] = c2[r];
        orow[48] = c3[r];
    }
}

// ---------------- Depthwise causal conv (K=4) + bias + SiLU ----------------
__global__ void conv_silu_kernel(const float* __restrict__ xz,
                                 const float* __restrict__ cw,
                                 const float* __restrict__ cb,
                                 float* __restrict__ u, int total) {
    int id = blockIdx.x * blockDim.x + threadIdx.x;
    if (id >= total) return;
    int c = id & (DIi - 1);
    int t = id >> 9;          // token b*L + l
    int l = t & (Ll - 1);
    const float* w = cw + c * 4;
    float acc = 0.f;
    #pragma unroll
    for (int k = 0; k < 4; ++k) {
        int ls = l - 3 + k;
        if (ls >= 0) acc += xz[(size_t)(t - 3 + k) * (2 * DIi) + c] * w[k];
    }
    acc += cb[c];
    u[id] = acc * sigmoidf_(acc);
}

// ---------------- dbl[NT,48] = u[NT,512] @ xp_w[48,512]^T ----------------
__global__ void xproj_kernel(const float* __restrict__ u,
                             const float* __restrict__ xp_w,
                             float* __restrict__ dbl, int total) {
    int id = blockIdx.x * blockDim.x + threadIdx.x;
    if (id >= total) return;
    int col = id % 48;
    int row = id / 48;
    const float* ur = u + (size_t)row * DIi;
    const float* wr = xp_w + (size_t)col * DIi;
    float acc = 0.f;
    #pragma unroll 8
    for (int k = 0; k < DIi; ++k) acc += ur[k] * wr[k];
    dbl[id] = acc;
}

// ---------------- dt[NT,512] = softplus(dbl[:, :16] @ dt_w^T + dt_b) ----------------
__global__ void dt_kernel(const float* __restrict__ dbl,
                          const float* __restrict__ dt_w,
                          const float* __restrict__ dt_b,
                          float* __restrict__ dtb, int total) {
    int id = blockIdx.x * blockDim.x + threadIdx.x;
    if (id >= total) return;
    int d = id & (DIi - 1);
    int t = id >> 9;
    const float* dr = dbl + (size_t)t * 48;
    const float* wr = dt_w + (size_t)d * Rr;
    float acc = dt_b[d];
    #pragma unroll
    for (int r = 0; r < Rr; ++r) acc += dr[r] * wr[r];
    dtb[id] = (acc > 20.f) ? acc : log1pf(expf(acc));
}

// ---------------- Selective scan + D-skip + SiLU(z) gating -> g (f16) ----------------
__global__ void scan_kernel(const float* __restrict__ dtb,
                            const float* __restrict__ u,
                            const float* __restrict__ dbl,
                            const float* __restrict__ xz,
                            const float* __restrict__ A_log,
                            const float* __restrict__ D_skip,
                            _Float16* __restrict__ g16, int total) {
    int id = blockIdx.x * blockDim.x + threadIdx.x;
    if (id >= total) return;
    int d = id & (DIi - 1);
    int b = id >> 9;
    float Ax[Nn], h[Nn];
    #pragma unroll
    for (int n = 0; n < Nn; ++n) {
        Ax[n] = -expf(A_log[(size_t)d * Nn + n]);
        h[n] = 0.f;
    }
    float ds = D_skip[d];
    for (int l = 0; l < Ll; ++l) {
        int t = b * Ll + l;
        float dtv = dtb[(size_t)t * DIi + d];
        float uv  = u[(size_t)t * DIi + d];
        float du = dtv * uv;
        const float* bc = dbl + (size_t)t * 48;
        float y = 0.f;
        #pragma unroll
        for (int n = 0; n < Nn; ++n) {
            h[n] = expf(dtv * Ax[n]) * h[n] + du * bc[Rr + n];
            y += h[n] * bc[Rr + Nn + n];
        }
        y += uv * ds;
        float zv = xz[(size_t)t * (2 * DIi) + DIi + d];
        float g = y * (zv * sigmoidf_(zv));
        g16[(size_t)t * DIi + d] = (_Float16)g;
    }
}

// ---------------- f32 -> f16 convert ----------------
__global__ void cvt_f32_f16(const float* __restrict__ src, _Float16* __restrict__ dst, int n) {
    int i = blockIdx.x * blockDim.x + threadIdx.x;
    if (i < n) dst[i] = (_Float16)src[i];
}

// ---------------- Final LN + classifier + halt head ----------------
__global__ void head_kernel(const float* __restrict__ feats,
                            const float* __restrict__ ng, const float* __restrict__ nb,
                            const float* __restrict__ cls_w, const float* __restrict__ cls_b,
                            const float* __restrict__ halt_w, const float* __restrict__ halt_b,
                            float* __restrict__ out) {
    __shared__ float red[256];
    int t = blockIdx.x;
    int tid = threadIdx.x;
    float v = feats[(size_t)t * Dd + tid];
    float mu  = block_reduce_256(red, tid, v) * (1.f / 256.f);
    float ex2 = block_reduce_256(red, tid, v * v) * (1.f / 256.f);
    float var = ex2 - mu * mu;
    float h = (v - mu) * rsqrtf(var + 1e-5f) * ng[tid] + nb[tid];
    float l0 = block_reduce_256(red, tid, h * cls_w[tid]);
    float l1 = block_reduce_256(red, tid, h * cls_w[Dd + tid]);
    float hp = block_reduce_256(red, tid, h * halt_w[tid]);
    if (tid == 0) {
        out[(size_t)t * 2 + 0] = l0 + cls_b[0];
        out[(size_t)t * 2 + 1] = l1 + cls_b[1];
        out[(size_t)NT * 2 + t] = sigmoidf_(hp + halt_b[0]);
    }
}

extern "C" void kernel_launch(void* const* d_in, const int* in_sizes, int n_in,
                              void* d_out, int out_size, void* d_ws, size_t ws_size,
                              hipStream_t stream) {
    const float* x         = (const float*)d_in[0];
    const float* emb_proto = (const float*)d_in[1];
    const float* emb_flags = (const float*)d_in[2];
    const float* emb_dir   = (const float*)d_in[3];
    const float* len_w     = (const float*)d_in[4];
    const float* len_b     = (const float*)d_in[5];
    const float* iat_w     = (const float*)d_in[6];
    const float* iat_b     = (const float*)d_in[7];
    const float* fus_w     = (const float*)d_in[8];
    const float* fus_b     = (const float*)d_in[9];
    const float* tok_g     = (const float*)d_in[10];
    const float* tok_b     = (const float*)d_in[11];
    const float* in_w      = (const float*)d_in[12];   // (3, 1024, 256)
    const float* conv_w    = (const float*)d_in[13];   // (3, 512, 1, 4)
    const float* conv_b    = (const float*)d_in[14];   // (3, 512)
    const float* xp_w      = (const float*)d_in[15];   // (3, 48, 512)
    const float* dt_w      = (const float*)d_in[16];   // (3, 512, 16)
    const float* dt_b      = (const float*)d_in[17];   // (3, 512)
    const float* A_log     = (const float*)d_in[18];   // (3, 512, 16)
    const float* D_skip    = (const float*)d_in[19];   // (3, 512)
    const float* out_w     = (const float*)d_in[20];   // (3, 256, 512)
    const float* norm_g    = (const float*)d_in[21];
    const float* norm_b    = (const float*)d_in[22];
    const float* cls_w     = (const float*)d_in[23];
    const float* cls_b     = (const float*)d_in[24];
    const float* halt_w    = (const float*)d_in[25];
    const float* halt_b    = (const float*)d_in[26];
    float* out = (float*)d_out;

    // Workspace carve-out (all L2-resident on MI455X; ~87 MB total)
    char* ws = (char*)d_ws;
    size_t off = 0;
    auto alloc = [&](size_t bytes) -> void* {
        off = (off + 255) & ~(size_t)255;
        void* p = ws + off;
        off += bytes;
        return p;
    };
    float*     feats   = (float*)alloc((size_t)NT * Dd * 4);
    _Float16*  feats16 = (_Float16*)alloc((size_t)NT * Dd * 2);
    float*     xz      = (float*)alloc((size_t)NT * 2 * DIi * 4);
    float*     u       = (float*)alloc((size_t)NT * DIi * 4);
    float*     dbl     = (float*)alloc((size_t)NT * 48 * 4);
    float*     dtb     = (float*)alloc((size_t)NT * DIi * 4);
    _Float16*  g16     = (_Float16*)alloc((size_t)NT * DIi * 2);
    _Float16*  w16in   = (_Float16*)alloc((size_t)2 * DIi * Dd * 2);
    _Float16*  w16out  = (_Float16*)alloc((size_t)Dd * DIi * 2);

    // 1) tokenize + LN
    tokenize_kernel<<<NT, 256, 0, stream>>>(x, emb_proto, emb_flags, emb_dir,
                                            len_w, len_b, iat_w, iat_b,
                                            fus_w, fus_b, tok_g, tok_b,
                                            feats, feats16);

    for (int layer = 0; layer < 3; ++layer) {
        const float* in_w_l   = in_w   + (size_t)layer * 2 * DIi * Dd;
        const float* conv_w_l = conv_w + (size_t)layer * DIi * 4;
        const float* conv_b_l = conv_b + (size_t)layer * DIi;
        const float* xp_w_l   = xp_w   + (size_t)layer * 48 * DIi;
        const float* dt_w_l   = dt_w   + (size_t)layer * DIi * Rr;
        const float* dt_b_l   = dt_b   + (size_t)layer * DIi;
        const float* A_log_l  = A_log  + (size_t)layer * DIi * Nn;
        const float* D_skip_l = D_skip + (size_t)layer * DIi;
        const float* out_w_l  = out_w  + (size_t)layer * Dd * DIi;

        // weight conversions to f16 for WMMA
        {
            int n = 2 * DIi * Dd;  // 262144
            cvt_f32_f16<<<(n + 255) / 256, 256, 0, stream>>>(in_w_l, w16in, n);
            n = Dd * DIi;          // 131072
            cvt_f32_f16<<<(n + 255) / 256, 256, 0, stream>>>(out_w_l, w16out, n);
        }

        // xz[NT,1024] = feats[NT,256] @ in_w^T   (WMMA, 16x64 strip per wave)
        {
            int waves = (NT / 16) * ((2 * DIi) / 64);   // 8192
            gemm_wmma_f16<<<(waves + 7) / 8, 256, 0, stream>>>(feats16, w16in, xz,
                                                               NT, 2 * DIi, Dd);
        }

        // depthwise conv + SiLU
        {
            int total = NT * DIi;
            conv_silu_kernel<<<(total + 255) / 256, 256, 0, stream>>>(xz, conv_w_l, conv_b_l,
                                                                      u, total);
        }

        // x-projection (48 outputs)
        {
            int total = NT * 48;
            xproj_kernel<<<(total + 255) / 256, 256, 0, stream>>>(u, xp_w_l, dbl, total);
        }

        // dt = softplus(...)
        {
            int total = NT * DIi;
            dt_kernel<<<(total + 255) / 256, 256, 0, stream>>>(dbl, dt_w_l, dt_b_l, dtb, total);
        }

        // selective scan + gating -> g16
        {
            int total = Bb * DIi;   // 8192 recurrences
            scan_kernel<<<(total + 255) / 256, 256, 0, stream>>>(dtb, u, dbl, xz,
                                                                 A_log_l, D_skip_l, g16, total);
        }

        // feats[NT,256] = g[NT,512] @ out_w^T   (WMMA, 16x64 strip per wave)
        {
            int waves = (NT / 16) * (Dd / 64);          // 2048
            gemm_wmma_f16<<<(waves + 7) / 8, 256, 0, stream>>>(g16, w16out, feats,
                                                               NT, Dd, DIi);
        }

        // feats -> f16 for next layer's GEMM
        {
            int n = NT * Dd;
            cvt_f32_f16<<<(n + 255) / 256, 256, 0, stream>>>(feats, feats16, n);
        }
    }

    // final LN + heads
    head_kernel<<<NT, 256, 0, stream>>>(feats, norm_g, norm_b, cls_w, cls_b,
                                        halt_w, halt_b, out);
}
